// EmbModule_17317308137990
// MI455X (gfx1250) — compile-verified, matrix operand order checked by reading, not run on hardware
//
#include <hip/hip_runtime.h>
#include <math.h>

typedef __attribute__((ext_vector_type(16))) _Float16 v16h;
typedef __attribute__((ext_vector_type(8)))  _Float16 v8h;
typedef __attribute__((ext_vector_type(8)))  float    v8f;

#define NUME_  100000
#define H_     4
#define DH_    32
#define K_     16
#define R_     17
#define DIN_   128
#define DT_    32
#define DINF_  160      /* DIN + DT */
#define N0_    131072
#define N1_    32768
#define N2_    4096
#define B_     2048
#define HIST_  2
#define LN_EPS_ 1e-5f

// ---------------------------------------------------------------- utilities
__device__ __forceinline__ float wave_sum32(float v) {
#pragma unroll
  for (int m = 1; m < 32; m <<= 1) v += __shfl_xor(v, m, 32);
  return v;
}
__device__ __forceinline__ float wave_max32(float v) {
#pragma unroll
  for (int m = 1; m < 32; m <<= 1) v = fmaxf(v, __shfl_xor(v, m, 32));
  return v;
}

// ---------------------------------------------------------------- zero init
__global__ void zero_kernel(float* __restrict__ p, int n) {
  int t = blockIdx.x * blockDim.x + threadIdx.x;
  if (t < n) p[t] = 0.0f;
}

// ------------------------------------------------- pack W into WMMA B layout
// Packed layout per relation: [kt(5)][nt(8)][lane(32)][elem(16)] halfs.
// B-operand (32x16 f16, KxN): lane l -> column n = nt*16 + (l&15), hs = l>>4;
// element e: v = e>>1, p = e&1; k = kt*32 + (v&4 ? 16 : 0) + hs*8 + (v&3)*2 + p
__global__ void pack_w_kernel(const float* __restrict__ W, _Float16* __restrict__ Wp,
                              int total) {
  int tid = blockIdx.x * blockDim.x + threadIdx.x;
  if (tid >= total) return;
  int e  = tid & 15;
  int l  = (tid >> 4) & 31;
  int nt = (tid >> 9) & 7;
  int kt = (tid >> 12) % 5;
  int r  = (tid >> 12) / 5;
  int hs = l >> 4;
  int n  = nt * 16 + (l & 15);
  int v  = e >> 1, p = e & 1;
  int k  = kt * 32 + ((v & 4) ? 16 : 0) + hs * 8 + (v & 3) * 2 + p;
  Wp[tid] = (_Float16)W[((size_t)r * DINF_ + k) * (H_ * DH_) + n];
}

// -------------------------------------- prep: gather + TimeEnc + LayerNorm
// mode 0: feat = emb[ids[row] % NUME]   (layer-0 input)
// mode 1: feat = relu(acc[row]/R)       (layer-1 input)
// One wave per row; lane handles elements j = lane + 32*s, s = 0..4.
__global__ void prep_kernel(const float* __restrict__ feat, int mode,
                            const int* __restrict__ ids,
                            const float* __restrict__ freq,
                            const float* __restrict__ phase,
                            const float* __restrict__ gamma,
                            const float* __restrict__ beta,
                            const int* __restrict__ ts_ptr,
                            _Float16* __restrict__ out, int N) {
  int wv = threadIdx.x >> 5, lane = threadIdx.x & 31;
  int row = blockIdx.x * 8 + wv;
  if (row >= N) return;
  int id = ids[row];
  int ts = *ts_ptr;
  float t = (float)(ts - id / NUME_);
  const float* base = (mode == 0) ? (feat + (size_t)(id % NUME_) * DIN_)
                                  : (feat + (size_t)row * DIN_);
  float x[5];
#pragma unroll
  for (int s = 0; s < 5; s++) {
    int j = lane + 32 * s;
    float v;
    if (j < DIN_) {
      v = base[j];
      if (mode == 1) v = fmaxf(v * (1.0f / R_), 0.0f);
    } else {
      int jj = j - DIN_;
      v = cosf(t * freq[jj] + phase[jj]);
    }
    x[s] = v;
  }
  float s1 = 0.0f, s2 = 0.0f;
#pragma unroll
  for (int s = 0; s < 5; s++) { s1 += x[s]; s2 += x[s] * x[s]; }
  s1 = wave_sum32(s1);
  s2 = wave_sum32(s2);
  float mu  = s1 * (1.0f / DINF_);
  float var = s2 * (1.0f / DINF_) - mu * mu;
  float inv = rsqrtf(var + LN_EPS_);
#pragma unroll
  for (int s = 0; s < 5; s++) {
    int j = lane + 32 * s;
    out[(size_t)row * DINF_ + j] = (_Float16)((x[s] - mu) * inv * gamma[j] + beta[j]);
  }
}

// --------------------------------------------------------- WMMA projection
// C[M,128] = A[M,160] (f16, row-major) x Wp (packed per-relation), f32 accum.
// Block = 256 threads = 8 waves; wave w owns N-tile w (N = 8*16 = 128).
// Each wave computes FOUR 16x16 M-tiles (block covers 64 rows): each B k-tile
// is loaded ONCE into registers and reused across the 4 A tiles (4x less B
// traffic from L2, 20 WMMAs per wave). 5 k-tiles of 32 cover K=160.
__device__ __forceinline__ v16h load_a_tile(const _Float16* __restrict__ arow,
                                            int kt, int hs) {
  // A 16x32 f16 tile: elems 0..7 = K contiguous at kt*32 + hs*8,
  //                   elems 8..15 = K contiguous at kt*32 + 16 + hs*8.
  v8h alo = *(const v8h*)(arow + kt * 32 + hs * 8);
  v8h ahi = *(const v8h*)(arow + kt * 32 + 16 + hs * 8);
  v16h a;
#pragma unroll
  for (int i = 0; i < 8; i++) { a[i] = alo[i]; a[8 + i] = ahi[i]; }
  return a;
}

__global__ void __launch_bounds__(256)
gemm_wmma_kernel(const _Float16* __restrict__ A,
                 const _Float16* __restrict__ Wp,
                 float* __restrict__ C, int M) {
  int wv = threadIdx.x >> 5, lane = threadIdx.x & 31;
  int hs = lane >> 4, mr = lane & 15;
  int m0 = blockIdx.x * 64;
  const _Float16* arow = A + (size_t)(m0 + mr) * DINF_;
  v8f c[4];
#pragma unroll
  for (int mt = 0; mt < 4; mt++) c[mt] = (v8f){};
#pragma unroll
  for (int kt = 0; kt < 5; kt++) {
    v16h b = *(const v16h*)(Wp + ((size_t)(kt * 8 + wv) * 32 + lane) * 16);
#pragma unroll
    for (int mt = 0; mt < 4; mt++) {
      v16h a = load_a_tile(arow + (size_t)mt * 16 * DINF_, kt, hs);
      c[mt] = __builtin_amdgcn_wmma_f32_16x16x32_f16(false, a, false, b,
                                                     (short)0, c[mt],
                                                     false, false);
    }
  }
  // C/D layout: VGPR j -> row hs*8 + j, col = lane&15 within tile.
#pragma unroll
  for (int mt = 0; mt < 4; mt++) {
    float* crow = C + (size_t)(m0 + mt * 16 + hs * 8) * 128 + wv * 16 + mr;
#pragma unroll
    for (int j = 0; j < 8; j++) crow[(size_t)j * 128] = c[mt][j];
  }
}

// ----------------------------------- per-row head dots: el = proj.al, er = proj.ar
// One wave per row; lane holds float4 (4 consecutive of 128); head = lane/8.
__global__ void eler_kernel(const float* __restrict__ proj,
                            const float* __restrict__ al,
                            const float* __restrict__ ar,
                            float* __restrict__ el, float* __restrict__ er,
                            int Nsrc, int Ndst) {
  int wv = threadIdx.x >> 5, lane = threadIdx.x & 31;
  int row = blockIdx.x * 8 + wv;
  if (row >= Nsrc) return;
  float4 p = *(const float4*)(proj + (size_t)row * 128 + lane * 4);
  float4 a = *(const float4*)(al + lane * 4);
  float pl = p.x * a.x + p.y * a.y + p.z * a.z + p.w * a.w;
  pl += __shfl_xor(pl, 1, 32); pl += __shfl_xor(pl, 2, 32); pl += __shfl_xor(pl, 4, 32);
  if ((lane & 7) == 0) el[(size_t)row * 4 + (lane >> 3)] = pl;
  if (row < Ndst) {
    float4 b = *(const float4*)(ar + lane * 4);
    float pr = p.x * b.x + p.y * b.y + p.z * b.z + p.w * b.w;
    pr += __shfl_xor(pr, 1, 32); pr += __shfl_xor(pr, 2, 32); pr += __shfl_xor(pr, 4, 32);
    if ((lane & 7) == 0) er[(size_t)row * 4 + (lane >> 3)] = pr;
  }
}

// ----------------------- edge softmax over K=16 + weighted aggregation (+bias)
// One wave per (dst i, head h). Lanes 0..15 compute logits; then every lane
// is one of the 32 head-dim outputs (coalesced proj gather across lanes).
__global__ void attn_kernel(const int* __restrict__ idx,
                            const float* __restrict__ el,
                            const float* __restrict__ er,
                            const float* __restrict__ proj,
                            const float* __restrict__ bias,
                            float* __restrict__ acc, int Ndst) {
  int wv = threadIdx.x >> 5, lane = threadIdx.x & 31;
  int g = blockIdx.x * 8 + wv;
  int i = g >> 2, h = g & 3;
  if (i >= Ndst) return;
  float erv = er[(size_t)i * 4 + h];
  int nid = 0;
  float e = -1e30f;
  if (lane < K_) {
    nid = idx[(size_t)i * K_ + lane];
    float ev = el[(size_t)nid * 4 + h] + erv;
    e = (ev > 0.0f) ? ev : 0.2f * ev;   // leaky_relu 0.2
  }
  float m  = wave_max32(e);
  float ex = (lane < K_) ? __expf(e - m) : 0.0f;
  float s  = wave_sum32(ex);
  float a  = ex / s;
  float out = bias[h * 32 + lane];
#pragma unroll
  for (int k = 0; k < K_; k++) {
    int   nk = __shfl(nid, k, 32);
    float ak = __shfl(a,   k, 32);
    out += ak * proj[(size_t)nk * 128 + h * 32 + lane];
  }
  acc[(size_t)i * 128 + h * 32 + lane] += out;
}

// --------------------------------------------- final: relu(acc1/R)[root_idx]
__global__ void final_kernel(const float* __restrict__ acc1,
                             const int* __restrict__ root,
                             float* __restrict__ out) {
  int g = blockIdx.x;       // B*HIST
  int d = threadIdx.x;      // 128
  int r = root[g];
  out[(size_t)g * 128 + d] = fmaxf(acc1[(size_t)r * 128 + d] * (1.0f / R_), 0.0f);
}

// ---------------------------------------------------------------------------
extern "C" void kernel_launch(void* const* d_in, const int* in_sizes, int n_in,
                              void* d_out, int out_size, void* d_ws, size_t ws_size,
                              hipStream_t stream) {
  (void)in_sizes; (void)n_in; (void)out_size; (void)ws_size;
  const float* emb    = (const float*)d_in[0];
  const float* freq   = (const float*)d_in[1];
  const float* phase  = (const float*)d_in[2];
  const float* gamma0 = (const float*)d_in[3];
  const float* beta0  = (const float*)d_in[4];
  const float* W0     = (const float*)d_in[5];
  const float* al0    = (const float*)d_in[6];
  const float* ar0    = (const float*)d_in[7];
  const float* bias0  = (const float*)d_in[8];
  const float* gamma1 = (const float*)d_in[9];
  const float* beta1  = (const float*)d_in[10];
  const float* W1     = (const float*)d_in[11];
  const float* al1    = (const float*)d_in[12];
  const float* ar1    = (const float*)d_in[13];
  const float* bias1  = (const float*)d_in[14];
  const int* ids0 = (const int*)d_in[15];
  const int* idx0 = (const int*)d_in[16];
  const int* ids1 = (const int*)d_in[17];
  const int* idx1 = (const int*)d_in[18];
  const int* root = (const int*)d_in[19];
  const int* tsp  = (const int*)d_in[20];

  char* ws = (char*)d_ws;
  size_t off = 0;
  auto alloc = [&](size_t bytes) -> char* {
    char* p = ws + off;
    off += (bytes + 255) & ~(size_t)255;
    return p;
  };
  _Float16* hn0 = (_Float16*)alloc((size_t)N0_ * DINF_ * 2);
  _Float16* hn1 = (_Float16*)alloc((size_t)N1_ * DINF_ * 2);
  _Float16* Wp0 = (_Float16*)alloc((size_t)R_ * DINF_ * 128 * 2);
  _Float16* Wp1 = (_Float16*)alloc((size_t)R_ * DINF_ * 128 * 2);
  float* proj = (float*)alloc((size_t)N0_ * 128 * 4);
  float* el   = (float*)alloc((size_t)N0_ * 4 * 4);
  float* er   = (float*)alloc((size_t)N1_ * 4 * 4);
  float* acc0 = (float*)alloc((size_t)N1_ * 128 * 4);
  float* acc1 = (float*)alloc((size_t)N2_ * 128 * 4);

  zero_kernel<<<(N1_ * 128 + 255) / 256, 256, 0, stream>>>(acc0, N1_ * 128);
  zero_kernel<<<(N2_ * 128 + 255) / 256, 256, 0, stream>>>(acc1, N2_ * 128);

  const int wtot = R_ * DINF_ * 128;
  pack_w_kernel<<<(wtot + 255) / 256, 256, 0, stream>>>(W0, Wp0, wtot);
  pack_w_kernel<<<(wtot + 255) / 256, 256, 0, stream>>>(W1, Wp1, wtot);

  // ----- layer 0
  prep_kernel<<<N0_ / 8, 256, 0, stream>>>(emb, 0, ids0, freq, phase, gamma0,
                                           beta0, tsp, hn0, N0_);
  for (int r = 0; r < R_; r++) {
    gemm_wmma_kernel<<<N0_ / 64, 256, 0, stream>>>(
        hn0, Wp0 + (size_t)r * DINF_ * 128, proj, N0_);
    eler_kernel<<<N0_ / 8, 256, 0, stream>>>(proj, al0 + r * 128, ar0 + r * 128,
                                             el, er, N0_, N1_);
    attn_kernel<<<N1_ * 4 / 8, 256, 0, stream>>>(
        idx0 + (size_t)r * N1_ * K_, el, er, proj, bias0 + r * 128, acc0, N1_);
  }

  // ----- layer 1
  prep_kernel<<<N1_ / 8, 256, 0, stream>>>(acc0, 1, ids1, freq, phase, gamma1,
                                           beta1, tsp, hn1, N1_);
  for (int r = 0; r < R_; r++) {
    gemm_wmma_kernel<<<N1_ / 64, 256, 0, stream>>>(
        hn1, Wp1 + (size_t)r * DINF_ * 128, proj, N1_);
    eler_kernel<<<N1_ / 8, 256, 0, stream>>>(proj, al1 + r * 128, ar1 + r * 128,
                                             el, er, N1_, N2_);
    attn_kernel<<<N2_ * 4 / 8, 256, 0, stream>>>(
        idx1 + (size_t)r * N2_ * K_, el, er, proj, bias1 + r * 128, acc1, N2_);
  }

  final_kernel<<<B_ * HIST_, 128, 0, stream>>>(acc1, root, (float*)d_out);
}
